// AttentionBlock_23330262352564
// MI455X (gfx1250) — matrix operand order repaired and checked by reading, not compiled, not run on hardware
//
#include <hip/hip_runtime.h>

// ---------- types for WMMA (CDNA5 gfx1250, wave32) ----------
typedef __attribute__((ext_vector_type(16))) __bf16 v16bf;
typedef __attribute__((ext_vector_type(8)))  __bf16 v8bf;
typedef __attribute__((ext_vector_type(8)))  float  v8f;

#define C_CH 512
#define LSEQ 1024
#define NB   8
#define NH   8
#define DH   64

__device__ __forceinline__ unsigned short f2bf(float f) {
    union { float f; unsigned u; } v; v.f = f;
    unsigned r = v.u + 0x7FFFu + ((v.u >> 16) & 1u);   // round-to-nearest-even
    return (unsigned short)(r >> 16);
}

// B operand: one contiguous 32B run (col-major memory, K contiguous per column)
__device__ __forceinline__ v16bf ldB(const unsigned short* p) {
    return *reinterpret_cast<const v16bf*>(p);
}
// A operand: two 16B runs (K 0-15 for lanes<16; K 16-31 for lanes>=16)
__device__ __forceinline__ v16bf ldA(const unsigned short* p0) {
    union { v16bf v; v8bf h[2]; } u;
    u.h[0] = *reinterpret_cast<const v8bf*>(p0);
    u.h[1] = *reinterpret_cast<const v8bf*>(p0 + 8);
    return u.v;
}
__device__ __forceinline__ v8f wmma_bf16(v16bf a, v16bf b, v8f c) {
    return __builtin_amdgcn_wmma_f32_16x16x32_bf16(false, a, false, b, (short)0, c,
                                                   false, false);
}
__device__ __forceinline__ float redmax16(float v) {
    v = fmaxf(v, __shfl_xor(v, 1, 32));
    v = fmaxf(v, __shfl_xor(v, 2, 32));
    v = fmaxf(v, __shfl_xor(v, 4, 32));
    v = fmaxf(v, __shfl_xor(v, 8, 32));
    return v;
}
__device__ __forceinline__ float redsum16(float v) {
    v += __shfl_xor(v, 1, 32);
    v += __shfl_xor(v, 2, 32);
    v += __shfl_xor(v, 4, 32);
    v += __shfl_xor(v, 8, 32);
    return v;
}

// ---------- 1) GroupNorm -> xn^T (b, L, C) bf16 ----------
__global__ void gn_kernel(const float* __restrict__ x, const float* __restrict__ g,
                          const float* __restrict__ be, unsigned short* __restrict__ xnT) {
    const int blk = blockIdx.x;            // 8*32 blocks
    const int b = blk >> 5, grp = blk & 31;
    const int cbase = grp * 16;
    const float* xb = x + (size_t)b * C_CH * LSEQ + (size_t)cbase * LSEQ;
    const int tid = threadIdx.x;
    float s = 0.f, s2 = 0.f;
    for (int c = 0; c < 16; ++c)
        for (int l = tid; l < LSEQ; l += 256) {
            float v = xb[c * LSEQ + l];
            s += v; s2 += v * v;
        }
    __shared__ float r1[256], r2[256];
    r1[tid] = s; r2[tid] = s2; __syncthreads();
    for (int off = 128; off > 0; off >>= 1) {
        if (tid < off) { r1[tid] += r1[tid + off]; r2[tid] += r2[tid + off]; }
        __syncthreads();
    }
    const float mean = r1[0] * (1.f / 16384.f);
    const float var  = r2[0] * (1.f / 16384.f) - mean * mean;
    const float rstd = rsqrtf(var + 1e-5f);
    for (int l = tid; l < LSEQ; l += 256) {
        unsigned short* o = xnT + (size_t)b * LSEQ * C_CH + (size_t)l * C_CH + cbase;
        for (int c = 0; c < 16; ++c) {
            float v = (xb[c * LSEQ + l] - mean) * rstd * g[cbase + c] + be[cbase + c];
            o[c] = f2bf(v);
        }
    }
}

// ---------- 2) weights -> bf16 (Wf = [Wq; Wkv] fused 1536x512, WoB 512x512) ----------
__global__ void wprep_kernel(const float* __restrict__ Wq, const float* __restrict__ Wkv,
                             const float* __restrict__ Wo, unsigned short* __restrict__ Wf,
                             unsigned short* __restrict__ WoB) {
    int i = blockIdx.x * 256 + threadIdx.x;
    const int NWF = 1536 * 512;
    if (i < NWF) {
        float v = (i < 512 * 512) ? Wq[i] : Wkv[i - 512 * 512];
        Wf[i] = f2bf(v);
    } else {
        int j = i - NWF;
        if (j < 512 * 512) WoB[j] = f2bf(Wo[j]);
    }
}

// ---------- 3) QKV projection: P(1536,1024) = Wf(1536,512) x xn(512,1024) ----------
// block = 256 thr (8 waves, 4Mx2N), wave tile 32x64, block tile 128x128
__global__ void proj_kernel(const unsigned short* __restrict__ Wf,
                            const unsigned short* __restrict__ xnT,
                            const float* __restrict__ bq, const float* __restrict__ bkv,
                            unsigned short* __restrict__ qT, unsigned short* __restrict__ kT,
                            unsigned short* __restrict__ vv) {
    const int b = blockIdx.z;
    const int lane = threadIdx.x & 31, wave = threadIdx.x >> 5;
    const int waveM = wave & 3, waveN = wave >> 2;
    const int m0 = blockIdx.x * 128 + waveM * 32;
    const int n0 = blockIdx.y * 128 + waveN * 64;
    const int ln = lane & 15, hi = lane >> 4;
    const unsigned short* Bt = xnT + (size_t)b * LSEQ * C_CH;

    v8f acc[2][4];
    v8f zero = {};
#pragma unroll
    for (int i = 0; i < 2; ++i)
#pragma unroll
        for (int j = 0; j < 4; ++j) acc[i][j] = zero;

    for (int kc = 0; kc < 512; kc += 32) {
        v16bf a[2], bb[4];
#pragma unroll
        for (int i = 0; i < 2; ++i)
            a[i] = ldA(Wf + (size_t)(m0 + i * 16 + ln) * 512 + kc + hi * 16);
#pragma unroll
        for (int j = 0; j < 4; ++j)
            bb[j] = ldB(Bt + (size_t)(n0 + j * 16 + ln) * 512 + kc + hi * 16);
#pragma unroll
        for (int i = 0; i < 2; ++i)
#pragma unroll
            for (int j = 0; j < 4; ++j) acc[i][j] = wmma_bf16(a[i], bb[j], acc[i][j]);
    }
#pragma unroll
    for (int i = 0; i < 2; ++i)
#pragma unroll
        for (int j = 0; j < 4; ++j)
#pragma unroll
            for (int e = 0; e < 8; ++e) {
                const int orow = m0 + i * 16 + e + hi * 8;     // fused output channel
                const int l    = n0 + j * 16 + ln;             // position
                float v = acc[i][j][e];
                if (orow < 512) {                              // q, pre-scaled by 1/dim_head
                    v = (v + bq[orow]) * 0.015625f;
                    const int hh = orow >> 6, d = orow & 63;
                    qT[(((size_t)b * NH + hh) * LSEQ + l) * DH + d] = f2bf(v);
                } else if (orow < 1024) {                      // k  -> (h, L, D)
                    const int r = orow - 512; v += bkv[r];
                    const int hh = r >> 6, d = r & 63;
                    kT[(((size_t)b * NH + hh) * LSEQ + l) * DH + d] = f2bf(v);
                } else {                                       // v  -> (h, D, L)
                    const int r = orow - 512; v += bkv[r];
                    const int rr = orow - 1024;
                    const int hh = rr >> 6, d = rr & 63;
                    vv[(((size_t)b * NH + hh) * DH + d) * LSEQ + l] = f2bf(v);
                }
            }
}

// ---------- 4) flash attention: per wave 16 query rows, s in chunks of 32 ----------
__global__ void attn_kernel(const unsigned short* __restrict__ qT,
                            const unsigned short* __restrict__ kT,
                            const unsigned short* __restrict__ vv,
                            unsigned short* __restrict__ aT) {
    const int h = blockIdx.y;                 // global head 0..63
    const int b = h >> 3, hh = h & 7;
    const int wave = threadIdx.x >> 5, lane = threadIdx.x & 31;
    const int ln = lane & 15, hi = lane >> 4;
    const int t0 = blockIdx.x * 128 + wave * 16;

    __shared__ unsigned short plds[8 * 16 * 32];      // per-wave 16x32 bf16 P tile
    unsigned short* myp = plds + wave * 512;

    const unsigned short* qh = qT + (size_t)h * LSEQ * DH;
    const unsigned short* kh = kT + (size_t)h * LSEQ * DH;
    const unsigned short* vh = vv + (size_t)h * DH * LSEQ;

    v16bf aq[2];
#pragma unroll
    for (int c = 0; c < 2; ++c)
        aq[c] = ldA(qh + (size_t)(t0 + ln) * DH + c * 32 + hi * 16);

    v8f zero = {};
    v8f acc[4];
#pragma unroll
    for (int n = 0; n < 4; ++n) acc[n] = zero;
    float m[8], lsum[8];
#pragma unroll
    for (int e = 0; e < 8; ++e) { m[e] = -3.0e30f; lsum[e] = 0.f; }

    for (int s0 = 0; s0 < LSEQ; s0 += 32) {
        v8f st[2]; st[0] = zero; st[1] = zero;
#pragma unroll
        for (int j = 0; j < 2; ++j)
#pragma unroll
            for (int c = 0; c < 2; ++c) {
                v16bf bk = ldB(kh + (size_t)(s0 + j * 16 + ln) * DH + c * 32 + hi * 16);
                st[j] = wmma_bf16(aq[c], bk, st[j]);
            }
        // online softmax (row = VGPR index within 16-lane half)
#pragma unroll
        for (int e = 0; e < 8; ++e) {
            float rmax = redmax16(fmaxf(st[0][e], st[1][e]));
            float mnew = fmaxf(m[e], rmax);
            float corr = __expf(m[e] - mnew);
            float p0 = __expf(st[0][e] - mnew);
            float p1 = __expf(st[1][e] - mnew);
            st[0][e] = p0; st[1][e] = p1;
            float rs = redsum16(p0 + p1);
            lsum[e] = lsum[e] * corr + rs;
            m[e] = mnew;
#pragma unroll
            for (int n = 0; n < 4; ++n) acc[n][e] = acc[n][e] * corr;
        }
        // D-layout -> A-layout re-swizzle through LDS
#pragma unroll
        for (int j = 0; j < 2; ++j)
#pragma unroll
            for (int e = 0; e < 8; ++e)
                myp[(e + hi * 8) * 32 + j * 16 + ln] = f2bf(st[j][e]);
        __syncthreads();
        v16bf pa = ldA(myp + ln * 32 + hi * 16);
#pragma unroll
        for (int n = 0; n < 4; ++n) {
            v16bf bv = ldB(vh + (size_t)(n * 16 + ln) * LSEQ + s0 + hi * 16);
            acc[n] = wmma_bf16(pa, bv, acc[n]);
        }
        __syncthreads();
    }
    // out^T (t, c) bf16 for the final GEMM's B operand
#pragma unroll
    for (int n = 0; n < 4; ++n)
#pragma unroll
        for (int e = 0; e < 8; ++e) {
            const int t = t0 + e + hi * 8;
            const int c = hh * DH + n * 16 + ln;
            float v = acc[n][e] / lsum[e];
            aT[((size_t)b * LSEQ + t) * C_CH + c] = f2bf(v);
        }
}

// ---------- 5) output projection + bias + residual (f32 out) ----------
__global__ void outproj_kernel(const unsigned short* __restrict__ WoB,
                               const unsigned short* __restrict__ aT,
                               const float* __restrict__ bo, const float* __restrict__ x,
                               float* __restrict__ out) {
    const int b = blockIdx.z;
    const int lane = threadIdx.x & 31, wave = threadIdx.x >> 5;
    const int waveM = wave & 3, waveN = wave >> 2;
    const int m0 = blockIdx.x * 128 + waveM * 32;
    const int n0 = blockIdx.y * 128 + waveN * 64;
    const int ln = lane & 15, hi = lane >> 4;
    const unsigned short* Bt = aT + (size_t)b * LSEQ * C_CH;

    v8f acc[2][4];
    v8f zero = {};
#pragma unroll
    for (int i = 0; i < 2; ++i)
#pragma unroll
        for (int j = 0; j < 4; ++j) acc[i][j] = zero;

    for (int kc = 0; kc < 512; kc += 32) {
        v16bf a[2], bb[4];
#pragma unroll
        for (int i = 0; i < 2; ++i)
            a[i] = ldA(WoB + (size_t)(m0 + i * 16 + ln) * 512 + kc + hi * 16);
#pragma unroll
        for (int j = 0; j < 4; ++j)
            bb[j] = ldB(Bt + (size_t)(n0 + j * 16 + ln) * 512 + kc + hi * 16);
#pragma unroll
        for (int i = 0; i < 2; ++i)
#pragma unroll
            for (int j = 0; j < 4; ++j) acc[i][j] = wmma_bf16(a[i], bb[j], acc[i][j]);
    }
#pragma unroll
    for (int i = 0; i < 2; ++i)
#pragma unroll
        for (int j = 0; j < 4; ++j)
#pragma unroll
            for (int e = 0; e < 8; ++e) {
                const int orow = m0 + i * 16 + e + hi * 8;
                const int l    = n0 + j * 16 + ln;
                const size_t idx = ((size_t)b * C_CH + orow) * LSEQ + l;
                out[idx] = acc[i][j][e] + bo[orow] + x[idx];
            }
}

extern "C" void kernel_launch(void* const* d_in, const int* in_sizes, int n_in,
                              void* d_out, int out_size, void* d_ws, size_t ws_size,
                              hipStream_t stream) {
    (void)in_sizes; (void)n_in; (void)out_size; (void)ws_size;
    const float* x     = (const float*)d_in[0];
    const float* gamma = (const float*)d_in[1];
    const float* beta  = (const float*)d_in[2];
    const float* Wq    = (const float*)d_in[3];
    const float* bq    = (const float*)d_in[4];
    const float* Wkv   = (const float*)d_in[5];
    const float* bkv   = (const float*)d_in[6];
    const float* Wo    = (const float*)d_in[7];
    const float* bo    = (const float*)d_in[8];

    char* w = (char*)d_ws;
    size_t off = 0;
    auto carve = [&](size_t bytes) -> void* {
        void* p = w + off;
        off = (off + bytes + 255) & ~(size_t)255;
        return p;
    };
    unsigned short* xnT = (unsigned short*)carve((size_t)NB * LSEQ * C_CH * 2);   // 8 MB
    unsigned short* Wf  = (unsigned short*)carve((size_t)1536 * 512 * 2);         // 1.5 MB
    unsigned short* WoB = (unsigned short*)carve((size_t)512 * 512 * 2);          // 0.5 MB
    unsigned short* qT  = (unsigned short*)carve((size_t)NB * NH * LSEQ * DH * 2);// 8 MB
    unsigned short* kT  = (unsigned short*)carve((size_t)NB * NH * LSEQ * DH * 2);
    unsigned short* vv  = (unsigned short*)carve((size_t)NB * NH * DH * LSEQ * 2);
    unsigned short* aT  = (unsigned short*)carve((size_t)NB * LSEQ * C_CH * 2);

    wprep_kernel<<<4096, 256, 0, stream>>>(Wq, Wkv, Wo, Wf, WoB);
    gn_kernel<<<NB * 32, 256, 0, stream>>>(x, gamma, beta, xnT);
    proj_kernel<<<dim3(12, 8, NB), 256, 0, stream>>>(Wf, xnT, bq, bkv, qT, kT, vv);
    attn_kernel<<<dim3(8, 64), 256, 0, stream>>>(qT, kT, vv, aT);
    outproj_kernel<<<dim3(4, 8, NB), 256, 0, stream>>>(WoB, aT, bo, x, (float*)d_out);
}
